// CausalSelfAttention_24893630447937
// MI455X (gfx1250) — matrix operand order repaired and checked by reading, not compiled, not run on hardware
//
#include <hip/hip_runtime.h>
#include <hip/hip_bf16.h>

#define DIM 1024
#define HD 128
#define NH 8
#define TSEQ 4096
#define ATTN_SCALE 0.1f
#define NEGV -1e30f

typedef __attribute__((ext_vector_type(16))) __bf16 bf16x16;
typedef __attribute__((ext_vector_type(8)))  float  f32x8;

union BF16F { bf16x16 v; unsigned short u[16]; };

__device__ __forceinline__ unsigned short f2bf_bits(float f) {
  union { float f; unsigned u; } a; a.f = f;
  unsigned r = a.u + 0x7FFFu + ((a.u >> 16) & 1u);   // round-to-nearest-even
  return (unsigned short)(r >> 16);
}

__device__ __forceinline__ f32x8 zero8() {
  f32x8 z = {0.f,0.f,0.f,0.f,0.f,0.f,0.f,0.f};
  return z;
}

// ---------------------------------------------------------------------------
// K1 / K4: C(MxN,f32) = A(MxK) * W(NxK,row-major,f32 scaled by lam)^T
// bf16 WMMA, 128x128 tile per block, BK=64, 8 waves each 32x64.
// ---------------------------------------------------------------------------
template <bool A_BF16>
__global__ __launch_bounds__(256)
void gemm_bf16_wmma(const void* __restrict__ Aany,
                    const float* __restrict__ W,
                    const float* __restrict__ lam_ptr, int lam_idx,
                    float* __restrict__ C,
                    int M, int N, int K) {
  __shared__ unsigned short As[128][68];
  __shared__ unsigned short Bs[128][68];
  const int tid  = threadIdx.x;
  const int w    = tid >> 5, lane = tid & 31;
  const int half = lane >> 4, ln = lane & 15;
  const int wm   = w & 3, wn = w >> 2;
  const int n0   = blockIdx.x * 128, m0 = blockIdx.y * 128;
  const float lam = lam_ptr[lam_idx];
  const float*          Af = (const float*)Aany;
  const unsigned short* Ab = (const unsigned short*)Aany;

  f32x8 acc[2][4];
  #pragma unroll
  for (int i = 0; i < 2; ++i)
    #pragma unroll
    for (int j = 0; j < 4; ++j) acc[i][j] = zero8();

  for (int kb = 0; kb < K; kb += 64) {
    if (kb + 64 < K) {
      __builtin_prefetch(&W[(size_t)n0 * K + kb + 64], 0, 1);
      if (A_BF16) __builtin_prefetch(&Ab[(size_t)m0 * K + kb + 64], 0, 1);
      else        __builtin_prefetch(&Af[(size_t)m0 * K + kb + 64], 0, 1);
    }
    #pragma unroll
    for (int i = 0; i < 32; ++i) {
      int idx = tid + i * 256;          // 0..8191
      int r = idx >> 6, c = idx & 63;
      if (A_BF16) As[r][c] = Ab[(size_t)(m0 + r) * K + kb + c];
      else        As[r][c] = f2bf_bits(Af[(size_t)(m0 + r) * K + kb + c]);
      Bs[r][c] = f2bf_bits(lam * W[(size_t)(n0 + r) * K + kb + c]);
    }
    __syncthreads();
    #pragma unroll
    for (int kc = 0; kc < 2; ++kc) {
      BF16F a[2], b[4];
      #pragma unroll
      for (int mt = 0; mt < 2; ++mt) {
        int m = 32 * wm + 16 * mt + ln;
        #pragma unroll
        for (int e = 0; e < 16; ++e) {
          int k = kc * 32 + (half << 4) + e;
          a[mt].u[e] = As[m][k];
        }
      }
      #pragma unroll
      for (int nt = 0; nt < 4; ++nt) {
        int n = 64 * wn + 16 * nt + ln;
        #pragma unroll
        for (int e = 0; e < 16; ++e) {
          int k = kc * 32 + (half << 4) + e;
          b[nt].u[e] = Bs[n][k];
        }
      }
      #pragma unroll
      for (int mt = 0; mt < 2; ++mt)
        #pragma unroll
        for (int nt = 0; nt < 4; ++nt)
          acc[mt][nt] = __builtin_amdgcn_wmma_f32_16x16x32_bf16(
              false, a[mt].v, false, b[nt].v, (short)0, acc[mt][nt], false, false);
    }
    __syncthreads();
  }
  #pragma unroll
  for (int mt = 0; mt < 2; ++mt)
    #pragma unroll
    for (int nt = 0; nt < 4; ++nt)
      #pragma unroll
      for (int v = 0; v < 8; ++v) {
        int row = m0 + 32 * wm + 16 * mt + v + 8 * half;
        int col = n0 + 64 * wn + 16 * nt + ln;
        C[(size_t)row * N + col] = acc[mt][nt][v];
      }
}

// ---------------------------------------------------------------------------
// K2: RMS-norm + RoPE + key_offset + ve-gate + attn-gate.  One block per (t,h).
// ---------------------------------------------------------------------------
__global__ __launch_bounds__(128)
void qkv_post_kernel(const float* __restrict__ qkv,
                     const float* __restrict__ x,
                     const float* __restrict__ ve,
                     const float* __restrict__ attn_gate_w,
                     const float* __restrict__ ve_gate_w,
                     const int* __restrict__ key_offset_p,
                     unsigned short* __restrict__ qb,
                     unsigned short* __restrict__ kb,
                     unsigned short* __restrict__ vb,
                     float* __restrict__ gate) {
  const int t = blockIdx.x, h = blockIdx.y, d = threadIdx.x;
  const int tp = (t > 0) ? t - 1 : 0;
  const int koff = *key_offset_p;
  __shared__ float red[128];
  __shared__ float qs[128], ks[128];
  __shared__ float gbc[1];

  const float qv  = qkv[(size_t)t  * 3 * DIM +           h * HD + d];
  const float kv  = qkv[(size_t)t  * 3 * DIM + DIM     + h * HD + d];
  const float kpv = qkv[(size_t)tp * 3 * DIM + DIM     + h * HD + d];
  const float vv  = qkv[(size_t)t  * 3 * DIM + 2 * DIM + h * HD + d];

  auto bsum = [&](float val) -> float {
    red[d] = val; __syncthreads();
    for (int s2 = 64; s2 > 0; s2 >>= 1) {
      if (d < s2) red[d] += red[d + s2];
      __syncthreads();
    }
    float r = red[0]; __syncthreads();
    return r;
  };
  const float eps = 1.1920929e-7f;
  float sq  = bsum(qv * qv);
  float sk  = bsum(kv * kv);
  float skp = bsum(kpv * kpv);
  float qn  = qv  * rsqrtf(sq  * (1.0f / HD) + eps);
  float kn  = kv  * rsqrtf(sk  * (1.0f / HD) + eps);
  float kpn = kpv * rsqrtf(skp * (1.0f / HD) + eps);

  qs[d] = qn; ks[d] = kn;
  if (d == 0) {
    float s1 = 0.f, s2 = 0.f;
    for (int j = 0; j < 6; ++j) {
      s1 += x [(size_t)t * DIM + j] * ve_gate_w[h * 12 + j];
      s1 += ve[(size_t)t * DIM + j] * ve_gate_w[h * 12 + 6 + j];
    }
    for (int j = 0; j < 12; ++j)
      s2 += x[(size_t)t * DIM + j] * attn_gate_w[h * 12 + j];
    gbc[0] = 2.0f / (1.0f + __expf(-s1));
    gate[(size_t)t * NH + h] = 1.0f / (1.0f + __expf(-s2));
  }
  __syncthreads();

  float qo, ko;
  if (d < 64) {
    int   i    = d >> 1;
    float freq = __powf(1.0f / 1024.0f, (float)i * (1.0f / 31.0f));
    float th   = (float)t * freq;
    float c = __cosf(th), s = __sinf(th);
    if ((d & 1) == 0) { qo = c * qs[d] + s * qs[d + 1]; ko = c * ks[d] + s * ks[d + 1]; }
    else              { qo = c * qs[d] - s * qs[d - 1]; ko = c * ks[d] - s * ks[d - 1]; }
  } else {
    qo = qn;
    ko = koff ? kpn : kn;   // af==0 on upper half -> rope is identity there
  }
  float vo = vv + gbc[0] * ve[(size_t)t * DIM + h * HD + d];
  size_t o = (size_t)t * DIM + h * HD + d;
  qb[o] = f2bf_bits(qo);
  kb[o] = f2bf_bits(ko);
  vb[o] = f2bf_bits(vo);
}

// ---------------------------------------------------------------------------
// K3: flash attention, one block per (128-query block, head), 8 waves.
// K tile staged via GLOBAL_LOAD_ASYNC_TO_LDS_B128 (ASYNCcnt), V via transpose.
// ---------------------------------------------------------------------------
__global__ __launch_bounds__(256)
void flash_attn_kernel(const unsigned short* __restrict__ qb,
                       const unsigned short* __restrict__ kb,
                       const unsigned short* __restrict__ vb,
                       const float* __restrict__ gate,
                       const int* __restrict__ seqlens, int nseq,
                       const int* __restrict__ bm_p,
                       unsigned short* __restrict__ yb) {
  __shared__ unsigned short Ks[64][136];     // [key][d], 272B rows: 16B-aligned
  __shared__ unsigned short Vt[128][68];     // [d][key]
  __shared__ unsigned short Ps[8][16][68];   // per-wave P strip

  const int tid  = threadIdx.x;
  const int w    = tid >> 5, lane = tid & 31;
  const int half = lane >> 4, ln = lane & 15;
  const int q0   = blockIdx.x * 128, h = blockIdx.y;
  const int bm   = *bm_p;

  // Q fragments for this wave's 16-row strip (rows q0+16w .. +15)
  BF16F qf[4];
  {
    const unsigned short* qrow = qb + (size_t)(q0 + 16 * w + ln) * DIM + h * HD;
    #pragma unroll
    for (int c = 0; c < 4; ++c)
      #pragma unroll
      for (int e = 0; e < 16; ++e) {
        int dd = 32 * c + (half << 4) + e;
        qf[c].u[e] = qrow[dd];
      }
  }
  const int qa0 = q0 + 16 * w + 8 * half;    // accumulator row v -> query qa0+v
  int docq[8];
  #pragma unroll
  for (int v = 0; v < 8; ++v) {
    int i = qa0 + v, dd = -1;
    for (int j = 0; j < nseq; ++j) dd += (seqlens[j] <= i) ? 1 : 0;
    docq[v] = dd;
  }
  float mrow[8], lrow[8];
  f32x8 oacc[8];
  #pragma unroll
  for (int v = 0; v < 8; ++v) { mrow[v] = NEGV; lrow[v] = 0.f; }
  #pragma unroll
  for (int nd = 0; nd < 8; ++nd) oacc[nd] = zero8();

  int kmin = q0 - bm; if (kmin < 0) kmin = 0;
  const int kt0 = kmin >> 6, kt1 = (q0 + 127) >> 6;
  for (int kt = kt0; kt <= kt1; ++kt) {
    const int kbase = kt * 64;

    // --- async DMA of the K tile: global -> LDS, 16B per lane per issue ---
    {
      const unsigned short* kg = kb + (size_t)kbase * DIM + h * HD;
      #pragma unroll
      for (int i = 0; i < 4; ++i) {
        int idx = tid + i * 256;               // 1024 chunks of 8 ushorts
        int r = idx >> 4, c16 = idx & 15;
        unsigned lds_a = (unsigned)(size_t)&Ks[r][c16 * 8];
        unsigned long long ga =
            (unsigned long long)(size_t)(kg + (size_t)r * DIM + c16 * 8);
        asm volatile("global_load_async_to_lds_b128 %0, %1, off"
                     :: "v"(lds_a), "v"(ga) : "memory");
      }
    }
    // --- V tile, transposed, regular path ---
    #pragma unroll
    for (int i = 0; i < 32; ++i) {
      int idx = tid + i * 256;                 // 64*128 elements
      int r = idx >> 7, c = idx & 127;
      Vt[c][r] = vb[(size_t)(kbase + r) * DIM + h * HD + c];
    }
    asm volatile("s_wait_asynccnt 0x0" ::: "memory");
    __syncthreads();

    // S = Q * K^T  (16 x 64 per wave)
    f32x8 s[4];
    #pragma unroll
    for (int n = 0; n < 4; ++n) {
      s[n] = zero8();
      #pragma unroll
      for (int c = 0; c < 4; ++c) {
        BF16F bfr;
        #pragma unroll
        for (int e = 0; e < 16; ++e)
          bfr.u[e] = Ks[16 * n + ln][32 * c + (half << 4) + e];
        s[n] = __builtin_amdgcn_wmma_f32_16x16x32_bf16(
            false, qf[c].v, false, bfr.v, (short)0, s[n], false, false);
      }
    }

    // mask + scale + online softmax
    float tmax[8];
    #pragma unroll
    for (int v = 0; v < 8; ++v) tmax[v] = NEGV;
    #pragma unroll
    for (int n = 0; n < 4; ++n) {
      int ka = kbase + 16 * n + ln;
      int dock = -1;
      for (int j = 0; j < nseq; ++j) dock += (seqlens[j] <= ka) ? 1 : 0;
      #pragma unroll
      for (int v = 0; v < 8; ++v) {
        int qa = qa0 + v;
        bool ok = (ka <= qa) && ((qa - ka) <= bm) && (dock == docq[v]);
        float sv = ok ? s[n][v] * ATTN_SCALE : NEGV;
        s[n][v] = sv;
        tmax[v] = fmaxf(tmax[v], sv);
      }
    }
    #pragma unroll
    for (int v = 0; v < 8; ++v)
      #pragma unroll
      for (int off = 1; off < 16; off <<= 1)
        tmax[v] = fmaxf(tmax[v], __shfl_xor(tmax[v], off, 32));

    float mnew[8], sc[8], tsum[8];
    #pragma unroll
    for (int v = 0; v < 8; ++v) {
      mnew[v] = fmaxf(mrow[v], tmax[v]);
      sc[v]   = __expf(mrow[v] - mnew[v]);
      tsum[v] = 0.f;
    }
    #pragma unroll
    for (int n = 0; n < 4; ++n)
      #pragma unroll
      for (int v = 0; v < 8; ++v) {
        float sv = s[n][v];
        float p  = (sv < -1e29f) ? 0.0f : __expf(sv - mnew[v]);
        s[n][v] = p;
        tsum[v] += p;
      }
    #pragma unroll
    for (int v = 0; v < 8; ++v) {
      #pragma unroll
      for (int off = 1; off < 16; off <<= 1)
        tsum[v] += __shfl_xor(tsum[v], off, 32);
      lrow[v] = lrow[v] * sc[v] + tsum[v];
      mrow[v] = mnew[v];
    }
    #pragma unroll
    for (int nd = 0; nd < 8; ++nd)
      #pragma unroll
      for (int v = 0; v < 8; ++v)
        oacc[nd][v] *= sc[v];

    // P -> per-wave LDS (wave-private; LDS is in-order within a wave)
    #pragma unroll
    for (int n = 0; n < 4; ++n)
      #pragma unroll
      for (int v = 0; v < 8; ++v)
        Ps[w][v + 8 * half][16 * n + ln] = f2bf_bits(s[n][v]);
    __builtin_amdgcn_wave_barrier();

    // O += P * V
    #pragma unroll
    for (int kc = 0; kc < 2; ++kc) {
      BF16F pa;
      #pragma unroll
      for (int e = 0; e < 16; ++e) {
        int key = 32 * kc + (half << 4) + e;
        pa.u[e] = Ps[w][ln][key];
      }
      #pragma unroll
      for (int nd = 0; nd < 8; ++nd) {
        BF16F vbf;
        #pragma unroll
        for (int e = 0; e < 16; ++e)
          vbf.u[e] = Vt[16 * nd + ln][32 * kc + (half << 4) + e];
        oacc[nd] = __builtin_amdgcn_wmma_f32_16x16x32_bf16(
            false, pa.v, false, vbf.v, (short)0, oacc[nd], false, false);
      }
    }
    __syncthreads();
  }

  // normalize, gate, store y (bf16)
  #pragma unroll
  for (int v = 0; v < 8; ++v) {
    int   qa = qa0 + v;
    float rl = (lrow[v] > 0.f) ? 1.0f / lrow[v] : 0.f;
    float g  = gate[(size_t)qa * NH + h];
    #pragma unroll
    for (int nd = 0; nd < 8; ++nd) {
      float val = oacc[nd][v] * rl * g;
      yb[(size_t)qa * DIM + h * HD + 16 * nd + ln] = f2bf_bits(val);
    }
  }
}

// ---------------------------------------------------------------------------
extern "C" void kernel_launch(void* const* d_in, const int* in_sizes, int n_in,
                              void* d_out, int out_size, void* d_ws, size_t ws_size,
                              hipStream_t stream) {
  const float* x           = (const float*)d_in[0];
  const float* ve          = (const float*)d_in[1];
  const float* qkvo_w      = (const float*)d_in[2];
  const float* sa_lambdas  = (const float*)d_in[3];
  const float* attn_gate_w = (const float*)d_in[4];
  const float* ve_gate_w   = (const float*)d_in[5];
  const int*   seqlens     = (const int*)d_in[6];
  const int*   bm_size     = (const int*)d_in[7];
  const int*   key_offset  = (const int*)d_in[8];
  const int    nseq        = in_sizes[6];

  char* ws = (char*)d_ws;
  float* qkv = (float*)ws;
  size_t off = (size_t)TSEQ * 3 * DIM * sizeof(float);
  unsigned short* qb = (unsigned short*)(ws + off); off += (size_t)TSEQ * DIM * 2;
  unsigned short* kb = (unsigned short*)(ws + off); off += (size_t)TSEQ * DIM * 2;
  unsigned short* vb = (unsigned short*)(ws + off); off += (size_t)TSEQ * DIM * 2;
  float* gateb = (float*)(ws + off);                off += (size_t)TSEQ * NH * sizeof(float);
  unsigned short* yb = (unsigned short*)(ws + off);

  // K1: qkv = x @ (lam0 * Wqkv)^T
  gemm_bf16_wmma<false><<<dim3(3 * DIM / 128, TSEQ / 128), 256, 0, stream>>>(
      x, qkvo_w, sa_lambdas, 0, qkv, TSEQ, 3 * DIM, DIM);

  // K2: rms/rope/key-offset/gates -> bf16 q,k,v + gate
  qkv_post_kernel<<<dim3(TSEQ, NH), 128, 0, stream>>>(
      qkv, x, ve, attn_gate_w, ve_gate_w, key_offset, qb, kb, vb, gateb);

  // K3: flash attention -> bf16 y
  flash_attn_kernel<<<dim3(TSEQ / 128, NH), 256, 0, stream>>>(
      qb, kb, vb, gateb, seqlens, nseq, bm_size, yb);

  // K4: out = y @ (lam1 * Wo)^T
  gemm_bf16_wmma<true><<<dim3(DIM / 128, TSEQ / 128), 256, 0, stream>>>(
      yb, qkvo_w + (size_t)3 * DIM * DIM, sa_lambdas, 1, (float*)d_out,
      TSEQ, DIM, DIM);
}
